// UnifiedMemoryBackbone_67723044324014
// MI455X (gfx1250) — matrix operand order repaired and checked by reading, not compiled, not run on hardware
//
#include <hip/hip_runtime.h>
#include <math.h>

// ---------------------------------------------------------------------------
// MI455X (gfx1250, wave32) implementation of the UnifiedMemoryBackbone step.
// Flash-attention-style retrieval over the 10000x768 memory bank using
// v_wmma_f32_16x16x32_bf16, followed by a WMMA fusion MLP + LayerNorm.
// Attention: 64-token workgroup tiles (512 threads / 16 waves); each memory
// B-fragment feeds 4 WMMAs -> ~4 GB total L2 traffic (binding resource).
// ---------------------------------------------------------------------------

typedef __bf16 bf16;
typedef bf16  v16bf __attribute__((ext_vector_type(16)));
typedef bf16  v8bf  __attribute__((ext_vector_type(8)));
typedef float v8f   __attribute__((ext_vector_type(8)));

#define DIM       768
#define MROWS     10000
#define CHUNK     256             // memory rows per chunk (16 tiles, 1/wave)
#define NCHUNK    40              // ceil(625/16)
#define MPAD      10240           // NCHUNK*CHUNK
#define NTILES    625             // MROWS / 16
#define TRA       64              // attention tokens per workgroup (4 row-tiles)
#define ABLK      128             // 8192 / TRA
#define TRF       32              // fusion tokens per workgroup (2 row-tiles)
#define FBLK      256             // 8192 / TRF
#define ATT_SCALE 0.03608439182435161f   // 1/sqrt(768)
#define LN_EPS    1e-5f

static_assert(NCHUNK * CHUNK == MPAD, "pad mismatch");

union ABu { v16bf v; v8bf h[2]; };

__device__ __forceinline__ v8f zero_v8f() {
  v8f z = {0.f,0.f,0.f,0.f,0.f,0.f,0.f,0.f};
  return z;
}

__device__ __forceinline__ v8f wmma_bf16(v16bf a, v16bf b, v8f c) {
  // D = A(16x32 bf16) * B(32x16 bf16) + C(16x16 f32)
  return __builtin_amdgcn_wmma_f32_16x16x32_bf16(
      false, a, false, b, (short)0, c, false, false);
}

// A-matrix fragment (16x32 bf16), row-major source with leading dim `ld`.
__device__ __forceinline__ v16bf load_a_frag(const bf16* base, int ld, int k0, int lane) {
  const int half = lane >> 4, m = lane & 15;
  const bf16* p = base + (size_t)m * ld + k0 + 8 * half;
  ABu u;
  u.h[0] = *(const v8bf*)(p);
  u.h[1] = *(const v8bf*)(p + 16);
  return u.v;
}

// B-matrix fragment (32x16 bf16) from transposed storage BT[n][k], leading dim `ld`.
__device__ __forceinline__ v16bf load_b_frag(const bf16* bt, int ld, int k0, int lane) {
  const int half = lane >> 4, n = lane & 15;
  const bf16* p = bt + (size_t)n * ld + k0 + 8 * half;
  ABu u;
  u.h[0] = *(const v8bf*)(p);
  u.h[1] = *(const v8bf*)(p + 16);
  return u.v;
}

__device__ __forceinline__ float rowmax16(float v) {
  v = fmaxf(v, __shfl_xor(v, 1, 16));
  v = fmaxf(v, __shfl_xor(v, 2, 16));
  v = fmaxf(v, __shfl_xor(v, 4, 16));
  v = fmaxf(v, __shfl_xor(v, 8, 16));
  return v;
}
__device__ __forceinline__ float rowsum16(float v) {
  v += __shfl_xor(v, 1, 16);
  v += __shfl_xor(v, 2, 16);
  v += __shfl_xor(v, 4, 16);
  v += __shfl_xor(v, 8, 16);
  return v;
}

// --------------------------- prep kernels ----------------------------------

__global__ void cvt_memB_kernel(const float* __restrict__ mem, bf16* __restrict__ memB) {
  int i = blockIdx.x * 256 + threadIdx.x;
  if (i >= MPAD * DIM) return;
  int row = i / DIM, d = i - row * DIM;
  float v = (row < MROWS) ? mem[(size_t)row * DIM + d] : 0.0f;
  memB[i] = (bf16)v;
}

__global__ void cvt_memT_kernel(const float* __restrict__ mem, bf16* __restrict__ memT) {
  int i = blockIdx.x * 256 + threadIdx.x;
  if (i >= DIM * MPAD) return;
  int d = i / MPAD, row = i - d * MPAD;
  float v = (row < MROWS) ? mem[(size_t)row * DIM + d] : 0.0f;
  memT[i] = (bf16)v;
}

__global__ void prep_w_kernel(const float* __restrict__ w1, const float* __restrict__ w2,
                              bf16* __restrict__ w1aT, bf16* __restrict__ w1bT,
                              bf16* __restrict__ w2T) {
  int i = blockIdx.x * 256 + threadIdx.x;
  if (i >= DIM * DIM) return;
  int n = i / DIM, k = i - n * DIM;   // BT[n][k]
  w1aT[i] = (bf16)(w1[(size_t)k * DIM + n] + w1[(size_t)(k + DIM) * DIM + n]);
  w1bT[i] = (bf16)(w1[(size_t)(k + 2 * DIM) * DIM + n]);
  w2T[i]  = (bf16)(w2[(size_t)k * DIM + n]);
}

// --------------------------- attention kernel ------------------------------
// grid = 128 (64-token tiles), block = 512 (16 waves).
// Wave w owns: logits col-tile (chunk tile c*16+w, all 4 row-tiles) and the
// output d-slice [w*48, w*48+48) for all 64 rows.

__global__ __launch_bounds__(512) void attn_kernel(
    const float* __restrict__ cs,
    const bf16*  __restrict__ memB,   // [MPAD][DIM]  bf16
    const bf16*  __restrict__ memT,   // [DIM][MPAD]  bf16
    const float* __restrict__ gate_w, // [2*DIM]
    const float* __restrict__ gate_b, // [1]
    bf16* __restrict__ xO,            // [8192][DIM]  bf16
    bf16* __restrict__ rO)            // [8192][DIM]  bf16
{
  __shared__ bf16  q_lds[TRA * DIM];     // 96 KB
  __shared__ bf16  p_lds[TRA * CHUNK];   // 32 KB
  __shared__ float wmax[16 * TRA];       // 4 KB
  __shared__ float wsum[16 * TRA];       // 4 KB
  __shared__ float mrun[TRA], lrun[TRA], alpha_s[TRA], mnew_s[TRA];
  __shared__ float gacc[TRA];

  const int tid   = threadIdx.x;
  const int w     = tid >> 5;       // 0..15
  const int lane  = tid & 31;
  const int half  = lane >> 4;
  const int lm    = lane & 15;
  const int t0    = blockIdx.x * TRA;
  const int dbase = w * 48;

  for (int i = tid; i < TRA * DIM; i += 512) {
    int row = i / DIM, d = i - row * DIM;
    q_lds[i] = (bf16)cs[(size_t)(t0 + row) * DIM + d];
  }
  if (tid < TRA) { mrun[tid] = -3.0e38f; lrun[tid] = 0.0f; }
  __syncthreads();

  v8f acc[4][3];
  #pragma unroll
  for (int g = 0; g < 4; ++g)
    #pragma unroll
    for (int n = 0; n < 3; ++n) acc[g][n] = zero_v8f();

  for (int c = 0; c < NCHUNK; ++c) {
    const int  tileIdx = c * 16 + w;
    const bool valid   = tileIdx < NTILES;     // uniform per wave
    float sv[4][8];
    #pragma unroll
    for (int g = 0; g < 4; ++g)
      #pragma unroll
      for (int r = 0; r < 8; ++r) sv[g][r] = 0.0f;

    // ---- logits: S[64,16] = Q[64,768] @ mem[tile]^T (B shared across g) ----
    if (valid) {
      const bf16* bp = memB + (size_t)(tileIdx * 16) * DIM;
      v8f s[4];
      #pragma unroll
      for (int g = 0; g < 4; ++g) s[g] = zero_v8f();
      #pragma unroll 4
      for (int kk = 0; kk < 24; ++kk) {
        v16bf b = load_b_frag(bp, DIM, kk * 32, lane);
        #pragma unroll
        for (int g = 0; g < 4; ++g) {
          v16bf a = load_a_frag(q_lds + g * 16 * DIM, DIM, kk * 32, lane);
          s[g] = wmma_bf16(a, b, s[g]);
        }
      }
      #pragma unroll
      for (int g = 0; g < 4; ++g)
        #pragma unroll
        for (int r = 0; r < 8; ++r) sv[g][r] = s[g][r] * ATT_SCALE;
    }

    // ---- per-row max across the wave's 16 columns ----
    #pragma unroll
    for (int g = 0; g < 4; ++g)
      #pragma unroll
      for (int r = 0; r < 8; ++r) {
        float m = valid ? rowmax16(sv[g][r]) : -3.0e38f;
        if (lm == 0) wmax[w * TRA + g * 16 + r + 8 * half] = m;
      }
    __syncthreads();

    if (w < 2) {              // waves 0,1 handle rows 0..31 / 32..63
      int row = w * 32 + lane;
      float cm = wmax[row];
      for (int j = 1; j < 16; ++j) cm = fmaxf(cm, wmax[j * TRA + row]);
      float mo = mrun[row];
      float mn = fmaxf(mo, cm);
      mnew_s[row]  = mn;
      alpha_s[row] = __expf(mo - mn);
    }
    __syncthreads();

    // ---- rescale running accumulator by alpha[row] ----
    #pragma unroll
    for (int g = 0; g < 4; ++g) {
      float al[8];
      #pragma unroll
      for (int r = 0; r < 8; ++r) al[r] = alpha_s[g * 16 + r + 8 * half];
      #pragma unroll
      for (int n = 0; n < 3; ++n)
        #pragma unroll
        for (int r = 0; r < 8; ++r) acc[g][n][r] *= al[r];
    }

    // ---- P = exp(S - m_new); stage in LDS; per-row partial sums ----
    #pragma unroll
    for (int g = 0; g < 4; ++g)
      #pragma unroll
      for (int r = 0; r < 8; ++r) {
        int row = g * 16 + r + 8 * half;
        float p = valid ? __expf(sv[g][r] - mnew_s[row]) : 0.0f;
        float ps = rowsum16(p);
        if (lm == 0) wsum[w * TRA + row] = ps;
        p_lds[row * CHUNK + w * 16 + lm] = (bf16)p;
      }
    __syncthreads();

    if (w < 2) {
      int row = w * 32 + lane;
      float s = 0.0f;
      for (int j = 0; j < 16; ++j) s += wsum[j * TRA + row];
      lrun[row] = lrun[row] * alpha_s[row] + s;
      mrun[row] = mnew_s[row];
    }

    // ---- O[64, 48(w)] += P[64,256] @ mem_chunk[256, 48(w)] (B shared x4) ----
    #pragma unroll
    for (int kk = 0; kk < 8; ++kk) {
      v16bf a0 = load_a_frag(p_lds,              CHUNK, kk * 32, lane);
      v16bf a1 = load_a_frag(p_lds + 16 * CHUNK, CHUNK, kk * 32, lane);
      v16bf a2 = load_a_frag(p_lds + 32 * CHUNK, CHUNK, kk * 32, lane);
      v16bf a3 = load_a_frag(p_lds + 48 * CHUNK, CHUNK, kk * 32, lane);
      #pragma unroll
      for (int n = 0; n < 3; ++n) {
        const bf16* bp = memT + (size_t)(dbase + n * 16) * MPAD;
        v16bf b = load_b_frag(bp, MPAD, c * CHUNK + kk * 32, lane);
        acc[0][n] = wmma_bf16(a0, b, acc[0][n]);
        acc[1][n] = wmma_bf16(a1, b, acc[1][n]);
        acc[2][n] = wmma_bf16(a2, b, acc[2][n]);
        acc[3][n] = wmma_bf16(a3, b, acc[3][n]);
      }
    }
    __syncthreads();   // protects p_lds / wmax / wsum for next chunk
  }

  // ---- normalize by softmax denominator ----
  #pragma unroll
  for (int g = 0; g < 4; ++g) {
    float inv[8];
    #pragma unroll
    for (int r = 0; r < 8; ++r) inv[r] = 1.0f / lrun[g * 16 + r + 8 * half];
    #pragma unroll
    for (int n = 0; n < 3; ++n)
      #pragma unroll
      for (int r = 0; r < 8; ++r) acc[g][n][r] *= inv[r];
  }

  // ---- long-term gate: sigmoid(cs.gw1 + ret.gw2 + b) ----
  if (tid < TRA) gacc[tid] = gate_b[0];
  __syncthreads();

  {
    float gpart[4][8];
    #pragma unroll
    for (int g = 0; g < 4; ++g)
      #pragma unroll
      for (int r = 0; r < 8; ++r) gpart[g][r] = 0.0f;
    #pragma unroll
    for (int n = 0; n < 3; ++n) {
      int d = dbase + n * 16 + lm;
      float g1 = gate_w[d];
      float g2 = gate_w[DIM + d];
      #pragma unroll
      for (int g = 0; g < 4; ++g)
        #pragma unroll
        for (int r = 0; r < 8; ++r) {
          int row = g * 16 + r + 8 * half;
          float cv = cs[(size_t)(t0 + row) * DIM + d];
          gpart[g][r] += cv * g1 + acc[g][n][r] * g2;
        }
    }
    #pragma unroll
    for (int g = 0; g < 4; ++g)
      #pragma unroll
      for (int r = 0; r < 8; ++r) {
        float s = rowsum16(gpart[g][r]);
        if (lm == 0) atomicAdd(&gacc[g * 16 + r + 8 * half], s);
      }
  }
  __syncthreads();

  // ---- x = cs + g*ret ; emit bf16 x and ret (re-read cs, cheap vs L2) ----
  #pragma unroll
  for (int g = 0; g < 4; ++g)
    #pragma unroll
    for (int r = 0; r < 8; ++r) {
      int row = g * 16 + r + 8 * half;
      float gt = 1.0f / (1.0f + __expf(-gacc[row]));
      #pragma unroll
      for (int n = 0; n < 3; ++n) {
        int d = dbase + n * 16 + lm;
        float rv = acc[g][n][r];
        float cv = cs[(size_t)(t0 + row) * DIM + d];
        float xv = cv + gt * rv;
        size_t o = (size_t)(t0 + row) * DIM + d;
        xO[o] = (bf16)xv;
        rO[o] = (bf16)rv;
      }
    }
}

// --------------------------- fusion MLP + LayerNorm ------------------------
// grid = 256 (32-token tiles), block = 256 (8 waves).
// Weight B-fragments shared across both row-tiles.

__global__ __launch_bounds__(256) void fusion_kernel(
    const bf16* __restrict__ xB,
    const bf16* __restrict__ rB,
    const bf16* __restrict__ w1aT,
    const bf16* __restrict__ w1bT,
    const bf16* __restrict__ w2T,
    const float* __restrict__ b1,
    const float* __restrict__ b2,
    const float* __restrict__ gamma,
    const float* __restrict__ beta,
    float* __restrict__ out)
{
  __shared__ bf16  h_lds[TRF * DIM];     // 48 KB
  __shared__ float ssum[TRF], ssq[TRF];

  const int tid   = threadIdx.x;
  const int w     = tid >> 5;
  const int lane  = tid & 31;
  const int half  = lane >> 4;
  const int lm    = lane & 15;
  const int t0    = blockIdx.x * TRF;
  const int dbase = w * 96;

  const bf16* xp = xB + (size_t)t0 * DIM;
  const bf16* rp = rB + (size_t)t0 * DIM;

  // ---- h = gelu(x@W1a + r@W1b + b1) ----
  v8f acc[2][6];
  #pragma unroll
  for (int g = 0; g < 2; ++g)
    #pragma unroll
    for (int n = 0; n < 6; ++n) acc[g][n] = zero_v8f();

  for (int kk = 0; kk < 24; ++kk) {
    v16bf ax0 = load_a_frag(xp,            DIM, kk * 32, lane);
    v16bf ax1 = load_a_frag(xp + 16 * DIM, DIM, kk * 32, lane);
    v16bf ar0 = load_a_frag(rp,            DIM, kk * 32, lane);
    v16bf ar1 = load_a_frag(rp + 16 * DIM, DIM, kk * 32, lane);
    #pragma unroll
    for (int n = 0; n < 6; ++n) {
      v16bf ba = load_b_frag(w1aT + (size_t)(dbase + n * 16) * DIM, DIM, kk * 32, lane);
      acc[0][n] = wmma_bf16(ax0, ba, acc[0][n]);
      acc[1][n] = wmma_bf16(ax1, ba, acc[1][n]);
      v16bf bb = load_b_frag(w1bT + (size_t)(dbase + n * 16) * DIM, DIM, kk * 32, lane);
      acc[0][n] = wmma_bf16(ar0, bb, acc[0][n]);
      acc[1][n] = wmma_bf16(ar1, bb, acc[1][n]);
    }
  }

  if (tid < TRF) { ssum[tid] = 0.0f; ssq[tid] = 0.0f; }

  #pragma unroll
  for (int n = 0; n < 6; ++n) {
    int d = dbase + n * 16 + lm;
    float bias = b1[d];
    #pragma unroll
    for (int g = 0; g < 2; ++g)
      #pragma unroll
      for (int r = 0; r < 8; ++r) {
        float v = acc[g][n][r] + bias;
        v = 0.5f * v * (1.0f + erff(v * 0.70710678118654752f));   // exact gelu
        h_lds[(g * 16 + r + 8 * half) * DIM + d] = (bf16)v;
      }
  }
  __syncthreads();

  // ---- o = h@W2 + b2 ----
  v8f acc2[2][6];
  #pragma unroll
  for (int g = 0; g < 2; ++g)
    #pragma unroll
    for (int n = 0; n < 6; ++n) acc2[g][n] = zero_v8f();

  for (int kk = 0; kk < 24; ++kk) {
    v16bf a0 = load_a_frag(h_lds,            DIM, kk * 32, lane);
    v16bf a1 = load_a_frag(h_lds + 16 * DIM, DIM, kk * 32, lane);
    #pragma unroll
    for (int n = 0; n < 6; ++n) {
      v16bf b = load_b_frag(w2T + (size_t)(dbase + n * 16) * DIM, DIM, kk * 32, lane);
      acc2[0][n] = wmma_bf16(a0, b, acc2[0][n]);
      acc2[1][n] = wmma_bf16(a1, b, acc2[1][n]);
    }
  }

  // ---- LayerNorm stats across d ----
  float s_r[2][8] = {{0,0,0,0,0,0,0,0},{0,0,0,0,0,0,0,0}};
  float q_r[2][8] = {{0,0,0,0,0,0,0,0},{0,0,0,0,0,0,0,0}};
  #pragma unroll
  for (int n = 0; n < 6; ++n) {
    int d = dbase + n * 16 + lm;
    float bias = b2[d];
    #pragma unroll
    for (int g = 0; g < 2; ++g)
      #pragma unroll
      for (int r = 0; r < 8; ++r) {
        float v = acc2[g][n][r] + bias;
        acc2[g][n][r] = v;
        s_r[g][r] += v;
        q_r[g][r] += v * v;
      }
  }
  #pragma unroll
  for (int g = 0; g < 2; ++g)
    #pragma unroll
    for (int r = 0; r < 8; ++r) {
      float s = rowsum16(s_r[g][r]);
      float q = rowsum16(q_r[g][r]);
      if (lm == 0) {
        atomicAdd(&ssum[g * 16 + r + 8 * half], s);
        atomicAdd(&ssq[g * 16 + r + 8 * half], q);
      }
    }
  __syncthreads();

  #pragma unroll
  for (int g = 0; g < 2; ++g)
    #pragma unroll
    for (int r = 0; r < 8; ++r) {
      int row = g * 16 + r + 8 * half;
      float mu  = ssum[row] * (1.0f / DIM);
      float var = ssq[row] * (1.0f / DIM) - mu * mu;
      float rs  = rsqrtf(var + LN_EPS);
      #pragma unroll
      for (int n = 0; n < 6; ++n) {
        int d = dbase + n * 16 + lm;
        out[(size_t)(t0 + row) * DIM + d] = (acc2[g][n][r] - mu) * rs * gamma[d] + beta[d];
      }
    }
}

// --------------------------- launch ----------------------------------------

extern "C" void kernel_launch(void* const* d_in, const int* in_sizes, int n_in,
                              void* d_out, int out_size, void* d_ws, size_t ws_size,
                              hipStream_t stream) {
  (void)in_sizes; (void)n_in; (void)out_size; (void)ws_size;
  const float* cs     = (const float*)d_in[0];
  const float* memory = (const float*)d_in[1];
  const float* gate_w = (const float*)d_in[2];
  const float* gate_b = (const float*)d_in[3];
  const float* w1     = (const float*)d_in[4];
  const float* b1     = (const float*)d_in[5];
  const float* w2     = (const float*)d_in[6];
  const float* b2     = (const float*)d_in[7];
  const float* gamma  = (const float*)d_in[8];
  const float* beta   = (const float*)d_in[9];

  char* ws = (char*)d_ws;
  bf16* memB = (bf16*)(ws + 0);          // MPAD*DIM*2  = 15,728,640
  bf16* memT = (bf16*)(ws + 15728640);   // DIM*MPAD*2  = 15,728,640
  bf16* w1aT = (bf16*)(ws + 31457280);   // DIM*DIM*2   =  1,179,648
  bf16* w1bT = (bf16*)(ws + 32636928);
  bf16* w2T  = (bf16*)(ws + 33816576);
  bf16* xB   = (bf16*)(ws + 34996224);   // 8192*DIM*2  = 12,582,912
  bf16* rB   = (bf16*)(ws + 47579136);   // total       = 60,162,048 bytes

  cvt_memB_kernel<<<(MPAD * DIM + 255) / 256, 256, 0, stream>>>(memory, memB);
  cvt_memT_kernel<<<(DIM * MPAD + 255) / 256, 256, 0, stream>>>(memory, memT);
  prep_w_kernel<<<(DIM * DIM + 255) / 256, 256, 0, stream>>>(w1, w2, w1aT, w1bT, w2T);

  attn_kernel<<<ABLK, 512, 0, stream>>>(cs, memB, memT, gate_w, gate_b, xB, rB);
  fusion_kernel<<<FBLK, 256, 0, stream>>>(xB, rB, w1aT, w1bT, w2T,
                                          b1, b2, gamma, beta, (float*)d_out);
}
